// LELayer_54022098649764
// MI455X (gfx1250) — matrix operand order repaired and checked by reading, not compiled, not run on hardware
//
#include <hip/hip_runtime.h>

// MI455X (gfx1250) fused KNN-weights layer.
//   xhi/xlo = bf16 split of x (x ~= hi + lo, error ~2^-24 rel)
//   sq[i]   = ||x_i||^2                      (f32)
//   xW      = x @ A                          (f32 WMMA 16x16x4, exact path)
//   fused:  s(i,j) = sq[j] - 2*dot(x_i,x_j)  via bf16x3 WMMA 16x16x32
//           top-10 per row in registers, out[i] = sum_t xW[idx[i][t]]
// Workspace: xW 32 MiB | sq 64 KiB | xhi 16 MiB | xlo 16 MiB.

typedef __attribute__((ext_vector_type(2)))  float v2f;
typedef __attribute__((ext_vector_type(8)))  float v8f;
typedef __attribute__((ext_vector_type(16))) __bf16 v16bf;
typedef __attribute__((ext_vector_type(8)))  unsigned short v8us;
typedef __attribute__((ext_vector_type(16))) unsigned short v16us;
typedef __attribute__((ext_vector_type(4)))  unsigned short v4us;
typedef int v4i_vs __attribute__((__vector_size__(16)));   // async-LDS builtin type

#define NROWS 16384
#define KDIM  512
#define ODIM  512
#define TOPK  10
#define PITCH 520      // LDS slab pitch (elems); 520*2/4 = 260 dwords == 4 mod 64

#define AS1 __attribute__((address_space(1)))
#define AS3 __attribute__((address_space(3)))

#if defined(__has_builtin)
#if __has_builtin(__builtin_amdgcn_global_load_async_to_lds_b128)
#define HAVE_ASYNC_LDS 1
#endif
#endif
#ifndef HAVE_ASYNC_LDS
#define HAVE_ASYNC_LDS 0
#endif

__device__ __forceinline__ v8f wmma_f32k4(v2f a, v2f b, v8f c) {
  return __builtin_amdgcn_wmma_f32_16x16x4_f32(false, a, false, b, (short)0, c,
                                               false, false);
}
__device__ __forceinline__ v8f wmma_bf16(v16bf a, v16bf b, v8f c) {
  return __builtin_amdgcn_wmma_f32_16x16x32_bf16(false, a, false, b, (short)0, c,
                                                 false, false);
}

// bf16 fragment loads (layouts per CDNA5 ISA 7.12.2, 16-bit A 16x32 / B 32x16)
union AFrag { struct { v8us c0, c1; } h; v16bf bf; };
union BFrag { v16us us; v16bf bf; };

__device__ __forceinline__ v16bf load_afrag(const unsigned short* row, int k, int hi) {
  AFrag f;
  f.h.c0 = *(const v8us*)(row + k + 8 * hi);        // VGPR0-3: K in [k+8hi, +8)
  f.h.c1 = *(const v8us*)(row + k + 16 + 8 * hi);   // VGPR4-7: K in [k+16+8hi, +8)
  return f.bf;
}
__device__ __forceinline__ v16bf load_bfrag(const unsigned short* row, int k, int hi) {
  AFrag f;
  f.h.c0 = *(const v8us*)(row + k + 8 * hi);        // K in [k+8hi, +8)
  f.h.c1 = *(const v8us*)(row + k + 16 + 8 * hi);   // K in [k+16+8hi, +8)
  return f.bf;
}

// ------------------- register-resident top-10 (smallest) -------------------
struct TopList {
  float v[TOPK];
  int   i[TOPK];
  float worst;
  int   wslot;
};
__device__ __forceinline__ void tl_init(TopList& t) {
  const float inf = __builtin_inff();
#pragma unroll
  for (int q = 0; q < TOPK; ++q) { t.v[q] = inf; t.i[q] = 0; }
  t.worst = inf; t.wslot = 0;
}
__device__ __forceinline__ void tl_insert(TopList& t, float s, int col) {
  if (s < t.worst) {
#pragma unroll
    for (int q = 0; q < TOPK; ++q)
      if (q == t.wslot) { t.v[q] = s; t.i[q] = col; }
    float w = t.v[0]; int ws = 0;
#pragma unroll
    for (int q = 1; q < TOPK; ++q)
      if (t.v[q] > w) { w = t.v[q]; ws = q; }
    t.worst = w; t.wslot = ws;
  }
}

// ------------------- kernel 0: f32 -> bf16 hi/lo split ---------------------
__device__ __forceinline__ unsigned short f2bf_rne(float f) {
  unsigned u = __float_as_uint(f);
  return (unsigned short)((u + 0x7fffu + ((u >> 16) & 1u)) >> 16);
}
__global__ __launch_bounds__(256)
void split_bf16_kernel(const float* __restrict__ x,
                       unsigned short* __restrict__ xhi,
                       unsigned short* __restrict__ xlo) {
  const int i = blockIdx.x * 256 + threadIdx.x;   // 4 elems per thread
  const float4 v = ((const float4*)x)[i];
  float f[4] = { v.x, v.y, v.z, v.w };
  v4us h, l;
#pragma unroll
  for (int c = 0; c < 4; ++c) {
    unsigned short hb = f2bf_rne(f[c]);
    float hf = __uint_as_float(((unsigned)hb) << 16);
    unsigned short lb = f2bf_rne(f[c] - hf);
    h[c] = hb; l[c] = lb;
  }
  *(v4us*)(xhi + 4 * (size_t)i) = h;
  *(v4us*)(xlo + 4 * (size_t)i) = l;
}

// ------------------- kernel 1: row squared norms ---------------------------
__global__ __launch_bounds__(256)
void rowsq_kernel(const float* __restrict__ x, float* __restrict__ sq) {
  const int row = blockIdx.x * 256 + threadIdx.x;
  const float4* p = (const float4*)(x + (size_t)row * KDIM);
  float acc = 0.0f;
#pragma unroll 8
  for (int i = 0; i < KDIM / 4; ++i) {
    float4 v = p[i];
    acc += v.x * v.x + v.y * v.y + v.z * v.z + v.w * v.w;
  }
  sq[row] = acc;
}

// ------------------- kernel 2: xW = x @ A  (exact f32 WMMA) ----------------
__global__ __launch_bounds__(256)
void xw_gemm_kernel(const float* __restrict__ x, const float* __restrict__ A,
                    float* __restrict__ xW) {
  const int lane = threadIdx.x & 31;
  const int wave = threadIdx.x >> 5;
  const int tile = blockIdx.x * 8 + wave;
  const int mBase = (tile >> 5) << 4;
  const int nBase = (tile & 31) << 4;
  const int lo = lane & 15, hi = lane >> 4;

  const float* aRow = x + (size_t)(mBase + lo) * KDIM;
  v8f acc = {};
#pragma unroll 4
  for (int k = 0; k < KDIM; k += 4) {
    const int k0 = k + 2 * hi;
    v2f av = *(const v2f*)(aRow + k0);
    v2f bv;
    bv.x = A[(size_t)k0 * ODIM + nBase + lo];
    bv.y = A[(size_t)(k0 + 1) * ODIM + nBase + lo];
    acc = wmma_f32k4(av, bv, acc);
  }
#pragma unroll
  for (int r = 0; r < 8; ++r)
    xW[(size_t)(mBase + r + 8 * hi) * ODIM + nBase + lo] = acc[r];
}

// ------------------- kernel 3: fused bf16x3 distances + top-10 + gather ----
// One block (8 waves) per 16-row slab; wave w covers column tiles
// [w*128, (w+1)*128) in pairs. K=512 in steps of 32 via WMMA bf16.
__global__ __launch_bounds__(256)
void knn_fused_kernel(const unsigned short* __restrict__ xhi,
                      const unsigned short* __restrict__ xlo,
                      const float* __restrict__ sq,
                      const float* __restrict__ xW,
                      float* __restrict__ out) {
  // [0,16384)        tileS  f32[8][16][32]
  // [16384,33024)    xhiR   u16[16][520]   (slab; dead after main loop)
  // [33024,49664)    xloR   u16[16][520]
  // overlay at 16384: candV f32[2560] | candI i32[2560] | fIdx i32[160]
  __shared__ __align__(16) char smem[49664];
  float*          tileS = (float*)smem;
  unsigned short* xhiR  = (unsigned short*)(smem + 16384);
  unsigned short* xloR  = (unsigned short*)(smem + 33024);
  float*          candV = (float*)(smem + 16384);
  int*            candI = (int*)(smem + 16384 + 10240);
  int*            fIdx  = (int*)(smem + 16384 + 20480);

  const int tid  = threadIdx.x;
  const int lane = tid & 31;
  const int wave = tid >> 5;
  const int lo = lane & 15, hi = lane >> 4;
  const int mBase = blockIdx.x << 4;
  const float inf = __builtin_inff();

  // ---- stage 16-row bf16 hi/lo slab into LDS (async DMA when available) ----
#if HAVE_ASYNC_LDS
  for (int c = tid; c < 2048; c += 256) {          // 2 arrays * 16 rows * 64 x 16B
    const int arr = c >> 10;
    const int r   = (c >> 6) & 15;
    const int off = (c & 63) * 16;                 // bytes within row
    const char* gsrc = (const char*)(arr ? xlo : xhi) +
                       (size_t)(mBase + r) * (KDIM * 2) + off;
    char* ldst = smem + 16384 + arr * 16640 + r * (PITCH * 2) + off;
    __builtin_amdgcn_global_load_async_to_lds_b128((AS1 v4i_vs*)gsrc,
                                                   (AS3 v4i_vs*)ldst, 0, 0);
  }
#if __has_builtin(__builtin_amdgcn_s_wait_asynccnt)
  __builtin_amdgcn_s_wait_asynccnt(0);
#else
  asm volatile("s_wait_asynccnt 0" ::: "memory");
#endif
#else
  for (int i = tid; i < 16 * KDIM; i += 256) {
    const int r = i >> 9, c = i & (KDIM - 1);
    xhiR[r * PITCH + c] = xhi[(size_t)(mBase + r) * KDIM + c];
    xloR[r * PITCH + c] = xlo[(size_t)(mBase + r) * KDIM + c];
  }
#endif
  __syncthreads();

  TopList list; tl_init(list);
  const unsigned short* aH = xhiR + lo * PITCH;
  const unsigned short* aL = xloR + lo * PITCH;

  for (int it = 0; it < 64; ++it) {
    const int nBase0 = (wave * 128 + it * 2) << 4;
    const int nBase1 = nBase0 + 16;
    const unsigned short* b0h = xhi + (size_t)(nBase0 + lo) * KDIM;
    const unsigned short* b0l = xlo + (size_t)(nBase0 + lo) * KDIM;
    const unsigned short* b1h = xhi + (size_t)(nBase1 + lo) * KDIM;
    const unsigned short* b1l = xlo + (size_t)(nBase1 + lo) * KDIM;

    v8f acc0 = {}, acc1 = {};
#pragma unroll 2
    for (int k = 0; k < KDIM; k += 32) {
      v16bf Ah  = load_afrag(aH, k, hi);           // LDS  2 x b128
      v16bf Al  = load_afrag(aL, k, hi);
      v16bf B0h = load_bfrag(b0h, k, hi);          // global 2 x b128
      v16bf B0l = load_bfrag(b0l, k, hi);
      v16bf B1h = load_bfrag(b1h, k, hi);
      v16bf B1l = load_bfrag(b1l, k, hi);
      // bf16x3: hi*hi + hi*lo + lo*hi  (~f32 accuracy; lo*lo ~2^-32 rel dropped)
      acc0 = wmma_bf16(Ah, B0h, acc0);
      acc0 = wmma_bf16(Ah, B0l, acc0);
      acc0 = wmma_bf16(Al, B0h, acc0);
      acc1 = wmma_bf16(Ah, B1h, acc1);
      acc1 = wmma_bf16(Ah, B1l, acc1);
      acc1 = wmma_bf16(Al, B1h, acc1);
    }

    const float sq0 = sq[nBase0 + lo];
    const float sq1 = sq[nBase1 + lo];
    float* ts = tileS + wave * 512;
#pragma unroll
    for (int r = 0; r < 8; ++r) {
      const int m  = r + 8 * hi;
      const int mg = mBase + m;
      float s0 = sq0 - 2.0f * acc0[r];
      float s1 = sq1 - 2.0f * acc1[r];
      if (mg == nBase0 + lo) s0 = inf;             // exclude diagonal
      if (mg == nBase1 + lo) s1 = inf;
      ts[m * 32 + lo]      = s0;
      ts[m * 32 + 16 + lo] = s1;
    }
    __syncthreads();

    const int row = lane >> 1, half = lane & 1;
    const float* rs = tileS + wave * 512 + row * 32 + half * 8;
#pragma unroll
    for (int j = 0; j < 8; ++j) tl_insert(list, rs[j],      nBase0 + half * 8 + j);
#pragma unroll
    for (int j = 0; j < 8; ++j) tl_insert(list, rs[16 + j], nBase1 + half * 8 + j);
  }
  __syncthreads();   // slab dead -> reuse for candidate merge

  {
    const int row = lane >> 1;
    const int slot = wave * 2 + (lane & 1);
#pragma unroll
    for (int q = 0; q < TOPK; ++q) {
      candV[row * 160 + slot * TOPK + q] = list.v[q];
      candI[row * 160 + slot * TOPK + q] = list.i[q];
    }
  }
  __syncthreads();

  if (tid < 16) {
    TopList fin; tl_init(fin);
    for (int s = 0; s < 160; ++s)
      tl_insert(fin, candV[tid * 160 + s], candI[tid * 160 + s]);
#pragma unroll
    for (int q = 0; q < TOPK; ++q) fIdx[tid * TOPK + q] = fin.i[q];
  }
  __syncthreads();

  for (int e = tid; e < 16 * ODIM; e += 256) {
    const int r = e >> 9, c = e & (ODIM - 1);
    float acc = 0.0f;
#pragma unroll
    for (int t = 0; t < TOPK; ++t)
      acc += xW[(size_t)fIdx[r * TOPK + t] * ODIM + c];
    out[(size_t)(mBase + r) * ODIM + c] = acc;
  }
}

// ---------------------------------------------------------------------------
extern "C" void kernel_launch(void* const* d_in, const int* in_sizes, int n_in,
                              void* d_out, int out_size, void* d_ws, size_t ws_size,
                              hipStream_t stream) {
  const float* x = (const float*)d_in[0];   // [16384, 512]
  const float* A = (const float*)d_in[1];   // [512, 512]
  float* out = (float*)d_out;               // [16384, 512]

  char* ws = (char*)d_ws;
  float*          xW  = (float*)ws;                                   // 32 MiB
  float*          sq  = (float*)(ws + (size_t)NROWS * ODIM * 4);      // 64 KiB
  unsigned short* xhi = (unsigned short*)(ws + (size_t)NROWS * ODIM * 4 + NROWS * 4);
  unsigned short* xlo = xhi + (size_t)NROWS * KDIM;

  split_bf16_kernel<<<NROWS * KDIM / 4 / 256, 256, 0, stream>>>(x, xhi, xlo);
  rowsq_kernel     <<<NROWS / 256, 256, 0, stream>>>(x, sq);
  xw_gemm_kernel   <<<(NROWS / 16) * (ODIM / 16) / 8, 256, 0, stream>>>(x, A, xW);
  knn_fused_kernel <<<NROWS / 16, 256, 0, stream>>>(xhi, xlo, sq, xW, out);
}
